// Block_59115929862601
// MI455X (gfx1250) — matrix operand order repaired and checked by reading, not compiled, not run on hardware
//
#include <hip/hip_runtime.h>
#include <hip/hip_bf16.h>
#include <math.h>

typedef __attribute__((ext_vector_type(16))) _Float16 v16h;
typedef __attribute__((ext_vector_type(8)))  _Float16 v8h;
typedef __attribute__((ext_vector_type(4)))  _Float16 v4h;
typedef __attribute__((ext_vector_type(8)))  float    v8f;

#define TILE_M 128
#define TILE_N 128
#define TILE_K 32
#define LDS_STRIDE 40   // 32 + 8 halves pad (80B rows, 16B-aligned subchunks)

// ---------------------------------------------------------------------------
// CDNA5 async memory->LDS staging (ASYNCcnt-tracked, no VGPR round trip).
// GVS mode: 64-bit SGPR base + per-lane 32-bit byte offset; vdst = LDS addr.
// ---------------------------------------------------------------------------
__device__ __forceinline__ void async_ld_b128(unsigned lds_byte_addr,
                                              unsigned long long gbase,
                                              unsigned gbyte_off)
{
    asm volatile("global_load_async_to_lds_b128 %0, %1, %2"
                 :: "v"(lds_byte_addr), "v"(gbyte_off), "s"(gbase)
                 : "memory");
}
__device__ __forceinline__ void wait_async0()
{
    asm volatile("s_wait_asynccnt 0" ::: "memory");
}
__device__ __forceinline__ unsigned lds_addr_u32(const void* p)
{
    return (unsigned)(uintptr_t)p;   // low 32 bits of flat LDS addr = LDS offset
}

// ---------------------------------------------------------------------------
// f16 WMMA GEMM: C(MxN) = A(MxK, row major) * Bt(NxK, row major == B^T).
// f32 accumulate, double-buffered LDS, async staging.
// EPI 0: store f32
// EPI 1: store f32 + bias[col] + res[row,col]
// EPI 2: t = tanh(acc*sc); store f16 Chebyshev T0..T3 at col*4+d
// EPI 3: store f32 acc*sc + res[row,col]
// ---------------------------------------------------------------------------
template<int EPI>
__global__ __launch_bounds__(256)
void gemm_wmma_f16(const _Float16* __restrict__ A,
                   const _Float16* __restrict__ Bt,
                   int M, int N, int K,
                   float* __restrict__ outF,
                   _Float16* __restrict__ outH,
                   const float* __restrict__ bias,
                   const float* __restrict__ res,
                   float sc)
{
    __shared__ __align__(16) _Float16 sA[2][TILE_M * LDS_STRIDE];
    __shared__ __align__(16) _Float16 sB[2][TILE_N * LDS_STRIDE]; // [n][k]

    const int tid  = threadIdx.x;
    const int lane = tid & 31;
    const int wave = tid >> 5;
    const int wm   = wave & 3;    // 4 waves along M (32 rows each)
    const int wn   = wave >> 2;   // 2 waves along N (64 cols each)
    const int ln   = lane & 15;
    const int lh   = lane >> 4;

    const int m0 = blockIdx.y * TILE_M;
    const int n0 = blockIdx.x * TILE_N;

    v8f acc[2][4];
#pragma unroll
    for (int i = 0; i < 2; ++i)
#pragma unroll
        for (int j = 0; j < 4; ++j)
#pragma unroll
            for (int r = 0; r < 8; ++r) acc[i][j][r] = 0.0f;

    // staging assignment: thread covers 16 halves (two b128 chunks) of one row
    const int srow = tid >> 1;            // 0..127
    const int scol = (tid & 1) * 16;      // 0 or 16
    const int gra  = m0 + srow;           // A global row (M dim)
    const int grb  = n0 + srow;           // Bt global row (N dim)
    const bool fullM = (m0 + TILE_M) <= M;

    const unsigned long long Ab  = (unsigned long long)(uintptr_t)A;
    const unsigned long long Btb = (unsigned long long)(uintptr_t)Bt;
    const unsigned aoff = ((unsigned)gra * (unsigned)K + (unsigned)scol) * 2u;
    const unsigned boff = ((unsigned)grb * (unsigned)K + (unsigned)scol) * 2u;

    unsigned sAaddr[2], sBaddr[2];
#pragma unroll
    for (int b = 0; b < 2; ++b) {
        sAaddr[b] = lds_addr_u32(&sA[b][srow * LDS_STRIDE + scol]);
        sBaddr[b] = lds_addr_u32(&sB[b][srow * LDS_STRIDE + scol]);
    }

    // ---- prologue: stage tile k0=0 into buffer 0 ----
    if (fullM) {
        async_ld_b128(sAaddr[0],      Ab, aoff);
        async_ld_b128(sAaddr[0] + 16, Ab, aoff + 16);
    } else {
        v8h a0, a1;
#pragma unroll
        for (int j = 0; j < 8; ++j) { a0[j] = (_Float16)0.0f; a1[j] = (_Float16)0.0f; }
        if (gra < M) {
            const v8h* gp = (const v8h*)(A + (size_t)gra * K + scol);
            a0 = gp[0]; a1 = gp[1];
        }
        *(v8h*)(&sA[0][srow * LDS_STRIDE + scol])     = a0;
        *(v8h*)(&sA[0][srow * LDS_STRIDE + scol + 8]) = a1;
    }
    async_ld_b128(sBaddr[0],      Btb, boff);
    async_ld_b128(sBaddr[0] + 16, Btb, boff + 16);
    wait_async0();
    __syncthreads();

    int pb = 0;
    for (int k0 = 0; k0 < K; k0 += TILE_K, pb ^= 1) {
        const bool hasNext = (k0 + TILE_K) < K;
        v8h a0, a1;                    // sync fallback regs (partial-M blocks)
        if (hasNext) {
            const unsigned kb = (unsigned)(k0 + TILE_K) * 2u;
            if (fullM) {
                async_ld_b128(sAaddr[pb ^ 1],      Ab, aoff + kb);
                async_ld_b128(sAaddr[pb ^ 1] + 16, Ab, aoff + kb + 16);
            } else {
#pragma unroll
                for (int j = 0; j < 8; ++j) { a0[j] = (_Float16)0.0f; a1[j] = (_Float16)0.0f; }
                if (gra < M) {
                    const v8h* gp = (const v8h*)(A + (size_t)gra * K + (k0 + TILE_K) + scol);
                    a0 = gp[0]; a1 = gp[1];
                }
            }
            async_ld_b128(sBaddr[pb ^ 1],      Btb, boff + kb);
            async_ld_b128(sBaddr[pb ^ 1] + 16, Btb, boff + kb + 16);
        }

        // ---- compute on buffer pb ----
        const _Float16* cA = sA[pb];
        const _Float16* cB = sB[pb];
        // B fragments: b[0..7] = B[lh*8+j][n], b[8..15] = B[16+lh*8+j][n]
        v16h bf[4];
#pragma unroll
        for (int nf = 0; nf < 4; ++nf) {
            const int nn = wn * 64 + nf * 16 + ln;
            v8h lo = *(const v8h*)(&cB[nn * LDS_STRIDE + lh * 8]);
            v8h hi = *(const v8h*)(&cB[nn * LDS_STRIDE + 16 + lh * 8]);
            bf[nf] = __builtin_shufflevector(lo, hi, 0,1,2,3,4,5,6,7,8,9,10,11,12,13,14,15);
        }
        // A fragments: a[0..7] = A[m][lh*8+j], a[8..15] = A[m][16+lh*8+j]
#pragma unroll
        for (int mf = 0; mf < 2; ++mf) {
            const int mm = wm * 32 + mf * 16 + ln;
            v8h lo = *(const v8h*)(&cA[mm * LDS_STRIDE + lh * 8]);
            v8h hi = *(const v8h*)(&cA[mm * LDS_STRIDE + 16 + lh * 8]);
            v16h af = __builtin_shufflevector(lo, hi, 0,1,2,3,4,5,6,7,8,9,10,11,12,13,14,15);
#pragma unroll
            for (int nf = 0; nf < 4; ++nf) {
                acc[mf][nf] = __builtin_amdgcn_wmma_f32_16x16x32_f16(
                    false, af, false, bf[nf], (short)0, acc[mf][nf], false, false);
            }
        }

        if (hasNext) {
            if (!fullM) {
                *(v8h*)(&sA[pb ^ 1][srow * LDS_STRIDE + scol])     = a0;
                *(v8h*)(&sA[pb ^ 1][srow * LDS_STRIDE + scol + 8]) = a1;
            }
            wait_async0();
        }
        __syncthreads();
    }

    // ---- epilogue: c[r] -> M = lh*8 + r, N = ln ----
    const bool needGuard = !fullM;
#pragma unroll
    for (int mf = 0; mf < 2; ++mf) {
#pragma unroll
        for (int nf = 0; nf < 4; ++nf) {
#pragma unroll
            for (int r = 0; r < 8; ++r) {
                const int row = m0 + wm * 32 + mf * 16 + lh * 8 + r;
                const int col = n0 + wn * 64 + nf * 16 + ln;
                if (!needGuard || row < M) {
                    float v = acc[mf][nf][r];
                    if constexpr (EPI == 0) {
                        outF[(size_t)row * N + col] = v;
                    } else if constexpr (EPI == 1) {
                        outF[(size_t)row * N + col] = v + bias[col] + res[(size_t)row * N + col];
                    } else if constexpr (EPI == 2) {
                        const float t  = tanhf(v * sc);
                        const float t2 = 2.0f * t * t - 1.0f;
                        const float t3 = t * (4.0f * t * t - 3.0f);
                        v4h o;
                        o[0] = (_Float16)1.0f; o[1] = (_Float16)t;
                        o[2] = (_Float16)t2;   o[3] = (_Float16)t3;
                        *(v4h*)(&outH[((size_t)row * N + col) * 4]) = o;
                    } else {
                        outF[(size_t)row * N + col] = v * sc + res[(size_t)row * N + col];
                    }
                }
            }
        }
    }
}

// ---------------------------------------------------------------------------
// LayerNorm -> f16 row
// ---------------------------------------------------------------------------
__global__ __launch_bounds__(256)
void ln_f16_kernel(const float* __restrict__ x, const float* __restrict__ g,
                   const float* __restrict__ b, _Float16* __restrict__ out, int C_)
{
    const int row = blockIdx.x;
    const int tid = threadIdx.x;
    const float* xr = x + (size_t)row * C_;
    float s = 0.f, sq = 0.f;
    for (int c = tid; c < C_; c += 256) { float v = xr[c]; s += v; sq += v * v; }
    __shared__ float red[16];
#pragma unroll
    for (int off = 16; off > 0; off >>= 1) {
        s  += __shfl_xor(s, off, 32);
        sq += __shfl_xor(sq, off, 32);
    }
    const int lane = tid & 31, wave = tid >> 5;
    if (lane == 0) { red[wave] = s; red[8 + wave] = sq; }
    __syncthreads();
    if (tid == 0) {
        float ts = 0.f, tq = 0.f;
        for (int i = 0; i < 8; ++i) { ts += red[i]; tq += red[8 + i]; }
        red[0] = ts; red[8] = tq;
    }
    __syncthreads();
    const float mean = red[0] / C_;
    const float var  = red[8] / C_ - mean * mean;
    const float inv  = rsqrtf(var + 1e-5f);
    for (int c = tid; c < C_; c += 256)
        out[(size_t)row * C_ + c] = (_Float16)((xr[c] - mean) * inv * g[c] + b[c]);
}

// LayerNorm -> tanh -> Chebyshev basis (4 f16 per element)
__global__ __launch_bounds__(256)
void ln_cheby_kernel(const float* __restrict__ x, const float* __restrict__ g,
                     const float* __restrict__ b, _Float16* __restrict__ basis, int C_)
{
    const int row = blockIdx.x;
    const int tid = threadIdx.x;
    const float* xr = x + (size_t)row * C_;
    float s = 0.f, sq = 0.f;
    for (int c = tid; c < C_; c += 256) { float v = xr[c]; s += v; sq += v * v; }
    __shared__ float red[16];
#pragma unroll
    for (int off = 16; off > 0; off >>= 1) {
        s  += __shfl_xor(s, off, 32);
        sq += __shfl_xor(sq, off, 32);
    }
    const int lane = tid & 31, wave = tid >> 5;
    if (lane == 0) { red[wave] = s; red[8 + wave] = sq; }
    __syncthreads();
    if (tid == 0) {
        float ts = 0.f, tq = 0.f;
        for (int i = 0; i < 8; ++i) { ts += red[i]; tq += red[8 + i]; }
        red[0] = ts; red[8] = tq;
    }
    __syncthreads();
    const float mean = red[0] / C_;
    const float var  = red[8] / C_ - mean * mean;
    const float inv  = rsqrtf(var + 1e-5f);
    for (int c = tid; c < C_; c += 256) {
        const float hn = (xr[c] - mean) * inv * g[c] + b[c];
        const float t  = tanhf(hn);
        const float t2 = 2.0f * t * t - 1.0f;
        const float t3 = t * (4.0f * t * t - 3.0f);
        v4h o;
        o[0] = (_Float16)1.0f; o[1] = (_Float16)t;
        o[2] = (_Float16)t2;   o[3] = (_Float16)t3;
        *(v4h*)(&basis[((size_t)row * C_ + c) * 4]) = o;
    }
}

// ---------------------------------------------------------------------------
// Flash-style attention: one wave per (b, h, query)
// ---------------------------------------------------------------------------
__global__ __launch_bounds__(256)
void attn_kernel(const float* __restrict__ qkv, _Float16* __restrict__ o16,
                 int Bb, int Nn, int Cc, int Hh)
{
    const int gw   = blockIdx.x * 8 + (threadIdx.x >> 5);
    const int lane = threadIdx.x & 31;
    const int n = gw % Nn;
    const int h = (gw / Nn) % Hh;
    const int b = gw / (Nn * Hh);
    const int hd = Cc / Hh;                 // 64 -> 2 floats per lane
    const size_t ld = (size_t)3 * Cc;

    const float2* qp = (const float2*)(qkv + ((size_t)(b * Nn + n)) * ld + h * hd);
    const float2 q = qp[lane];
    const float scale = rsqrtf((float)hd);

    const float* kbase = qkv + (size_t)b * Nn * ld + Cc + h * hd;
    const float* vbase = qkv + (size_t)b * Nn * ld + 2 * Cc + h * hd;

    float mx = -3.0e38f, sum = 0.f, o0 = 0.f, o1 = 0.f;
    for (int m = 0; m < Nn; ++m) {
        const float2 kv = ((const float2*)(kbase + (size_t)m * ld))[lane];
        float p = q.x * kv.x + q.y * kv.y;
#pragma unroll
        for (int off = 16; off > 0; off >>= 1) p += __shfl_xor(p, off, 32);
        p *= scale;
        const float nm   = fmaxf(mx, p);
        const float corr = __expf(mx - nm);
        const float e    = __expf(p - nm);
        const float2 vv  = ((const float2*)(vbase + (size_t)m * ld))[lane];
        sum = sum * corr + e;
        o0  = o0 * corr + e * vv.x;
        o1  = o1 * corr + e * vv.y;
        mx = nm;
    }
    const float invs = 1.0f / sum;
    _Float16* op = o16 + ((size_t)(b * Nn + n)) * Cc + h * hd + lane * 2;
    op[0] = (_Float16)(o0 * invs);
    op[1] = (_Float16)(o1 * invs);
    (void)Bb;
}

// ---------------------------------------------------------------------------
// weight preparation: cast + transpose to Bt (N x K) f16
// ---------------------------------------------------------------------------
__global__ __launch_bounds__(256)
void cast_f16_t_kernel(const float* __restrict__ s, _Float16* __restrict__ d,
                       int K, int N)   // s: K x N, d: N x K
{
    const int idx = blockIdx.x * 256 + threadIdx.x;
    if (idx >= K * N) return;
    const int k = idx / N;
    const int n = idx % N;
    d[(size_t)n * K + k] = (_Float16)s[idx];
}

// (I, O, 4) f32 -> transposed (O, 4I) f16, scaled: Wt[o][i*4+d] = c[i,o,d]*scale
__global__ __launch_bounds__(256)
void repack_cheby_t_kernel(const float* __restrict__ c, _Float16* __restrict__ w,
                           int I, int O, float scale)
{
    const int idx = blockIdx.x * 256 + threadIdx.x;
    if (idx >= I * O) return;
    const int i = idx % I;       // i fastest -> coalesced 8B writes
    const int o = idx / I;
    const float* src = c + ((size_t)i * O + o) * 4;
    v4h v;
#pragma unroll
    for (int d = 0; d < 4; ++d) v[d] = (_Float16)(src[d] * scale);
    *(v4h*)(&w[(size_t)o * (4 * I) + i * 4]) = v;
}

// ---------------------------------------------------------------------------
extern "C" void kernel_launch(void* const* d_in, const int* in_sizes, int n_in,
                              void* d_out, int out_size, void* d_ws, size_t ws_size,
                              hipStream_t stream)
{
    const float* x      = (const float*)d_in[0];
    const float* g1     = (const float*)d_in[1];
    const float* b1     = (const float*)d_in[2];
    const float* w_qkv  = (const float*)d_in[3];
    const float* w_proj = (const float*)d_in[4];
    const float* b_proj = (const float*)d_in[5];
    const float* g2     = (const float*)d_in[6];
    const float* b2     = (const float*)d_in[7];
    const float* c1     = (const float*)d_in[8];
    const float* c2     = (const float*)d_in[9];
    float* out = (float*)d_out;

    constexpr int B = 32, N = 197, C = 768, HEADS = 12, H = 3072;
    constexpr int BN = B * N;                 // 6304
    constexpr float WSCALE = 64.0f;           // keep tiny cheby coeffs in f16 normal range
    constexpr float INV_WSCALE = 1.0f / 64.0f;

    char* ws = (char*)d_ws;
    size_t off = 0;
    auto alloc = [&](size_t bytes) { char* p = ws + off; off += (bytes + 255) & ~(size_t)255; return p; };

    _Float16* h16     = (_Float16*)alloc((size_t)BN * C * 2);
    _Float16* wqkv16t = (_Float16*)alloc((size_t)C * 3 * C * 2);   // (3C x C)
    float*    qkv32   = (float*)   alloc((size_t)BN * 3 * C * 4);
    _Float16* o16     = (_Float16*)alloc((size_t)BN * C * 2);
    _Float16* wproj16t= (_Float16*)alloc((size_t)C * C * 2);       // (C x C)
    float*    x1      = (float*)   alloc((size_t)BN * C * 4);
    _Float16* basis1  = (_Float16*)alloc((size_t)BN * 4 * C * 2);
    _Float16* w1_16t  = (_Float16*)alloc((size_t)4 * C * H * 2);   // (H x 4C)
    _Float16* basis2  = (_Float16*)alloc((size_t)BN * 4 * H * 2);
    _Float16* w2_16t  = (_Float16*)alloc((size_t)4 * H * C * 2);   // (C x 4H)
    (void)ws_size;

    const int MT = (BN + TILE_M - 1) / TILE_M;   // 50

    // weight preparation (transposed f16 so GEMM B staging is contiguous/async)
    cast_f16_t_kernel<<<(C * 3 * C + 255) / 256, 256, 0, stream>>>(w_qkv, wqkv16t, C, 3 * C);
    cast_f16_t_kernel<<<(C * C + 255) / 256, 256, 0, stream>>>(w_proj, wproj16t, C, C);
    repack_cheby_t_kernel<<<(C * H + 255) / 256, 256, 0, stream>>>(c1, w1_16t, C, H, WSCALE);
    repack_cheby_t_kernel<<<(H * C + 255) / 256, 256, 0, stream>>>(c2, w2_16t, H, C, WSCALE);

    // 1) LN1 -> h16
    ln_f16_kernel<<<BN, 256, 0, stream>>>(x, g1, b1, h16, C);

    // 2) qkv = h @ w_qkv   (6304 x 2304 x 768)
    gemm_wmma_f16<0><<<dim3(3 * C / TILE_N, MT), 256, 0, stream>>>(
        h16, wqkv16t, BN, 3 * C, C, qkv32, nullptr, nullptr, nullptr, 1.0f);

    // 3) attention -> o16
    attn_kernel<<<(B * HEADS * N) / 8, 256, 0, stream>>>(qkv32, o16, B, N, C, HEADS);

    // 4) x1 = x + o @ w_proj + b_proj   (6304 x 768 x 768)
    gemm_wmma_f16<1><<<dim3(C / TILE_N, MT), 256, 0, stream>>>(
        o16, wproj16t, BN, C, C, x1, nullptr, b_proj, x, 1.0f);

    // 5) LN2 + cheby basis -> basis1 (6304 x 3072 f16)
    ln_cheby_kernel<<<BN, 256, 0, stream>>>(x1, g2, b2, basis1, C);

    // 6) h_mid = basis1 @ W1, fused tanh+cheby -> basis2 (6304 x 12288 f16)
    gemm_wmma_f16<2><<<dim3(H / TILE_N, MT), 256, 0, stream>>>(
        basis1, w1_16t, BN, H, 4 * C, nullptr, basis2, nullptr, nullptr, INV_WSCALE);

    // 7) out = x1 + basis2 @ W2   (6304 x 768 x 12288)
    gemm_wmma_f16<3><<<dim3(C / TILE_N, MT), 256, 0, stream>>>(
        basis2, w2_16t, BN, C, 4 * H, out, nullptr, nullptr, x1, INV_WSCALE);

    (void)in_sizes; (void)n_in; (void)out_size;
}
